// MultiDilatelocalAttention_47390669144854
// MI455X (gfx1250) — matrix operand order repaired and checked by reading, not compiled, not run on hardware
//
#include <hip/hip_runtime.h>
#include <hip/hip_bf16.h>
#include <math.h>

// ---------------------------------------------------------------------------
// MultiDilateLocalAttention for MI455X (gfx1250, wave32, WMMA bf16)
//   B=8, C=256, H=W=64, L=4096, ND=4 dilations (1,2,4,8), heads=2, head_dim=32
// ---------------------------------------------------------------------------

typedef __bf16 bf16_t;
typedef __attribute__((ext_vector_type(16))) __bf16 v16bf;
typedef __attribute__((ext_vector_type(8)))  __bf16 v8bf;
typedef __attribute__((ext_vector_type(8)))  float  v8f;

#define C_DIM   256
#define L_PIX   4096
#define BATCH   8
#define SCALE_F 0.17677669529663687f   // 32^-0.5
#define LN_EPS  1e-5f

// ---------------------------------------------------------------------------
// x (B,C,L) f32  ->  xT (B,L,C) bf16   (K-contiguous activation for WMMA B)
// ---------------------------------------------------------------------------
__global__ void k_x_to_bf16T(const float* __restrict__ x, bf16_t* __restrict__ xT) {
    int idx = blockIdx.x * 256 + threadIdx.x;     // over B*L*C = 8,388,608
    int c = idx & 255;
    int l = (idx >> 8) & (L_PIX - 1);
    int b = idx >> 20;
    xT[idx] = (bf16_t)x[((long)b * C_DIM + c) * L_PIX + l];
}

__global__ void k_f32_to_bf16(const float* __restrict__ in, bf16_t* __restrict__ out, int n) {
    int i = blockIdx.x * 256 + threadIdx.x;
    if (i < n) out[i] = (bf16_t)in[i];
}

__device__ __forceinline__ v16bf frag16(v8bf lo, v8bf hi) {
    v16bf r;
#pragma unroll
    for (int i = 0; i < 8; ++i) { r[i] = lo[i]; r[i + 8] = hi[i]; }
    return r;
}

// ---------------------------------------------------------------------------
// bf16 WMMA GEMM:  C[m][n] = sum_k A[m][k] * Bt[n][k]  + bias[m]
//   A  : M x K row-major bf16 (weights)
//   Bt : N x K row-major bf16 (activations, K contiguous), per-batch stride
//   C  : M x N row-major f32 (channel-major output), per-batch stride
// One wave -> 16(M) x 64(N) strip = 4 WMMA subtiles sharing one A fragment
// per k-step. All 10 b128 loads of a k-step are issued before the 4 WMMAs
// so the scheduler can use partial s_wait_loadcnt and overlap load/matrix.
// Block = 8 waves -> 16(M) x 512(N).
// ---------------------------------------------------------------------------
__global__ void k_gemm_bf16(const bf16_t* __restrict__ A, const bf16_t* __restrict__ Bt,
                            const float* __restrict__ bias, float* __restrict__ Cc,
                            int M, int N, int K, long bBatchStride, long cBatchStride) {
    int wave = threadIdx.x >> 5;
    int lane = threadIdx.x & 31;
    int half = lane >> 4;       // 0: lanes 0-15, 1: lanes 16-31
    int r    = lane & 15;

    int m0 = blockIdx.y * 16;
    int n0 = blockIdx.x * 512 + wave * 64;

    const bf16_t* Bb = Bt + (long)blockIdx.z * bBatchStride;
    float*        Cb = Cc + (long)blockIdx.z * cBatchStride;

    // A 16x32 layout (matches B): lane<16 holds M=lane, K=0..15;
    // lane>=16 holds M=lane-16, K=16..31.
    const bf16_t* ap = A  + (long)(m0 + r) * K + half * 16;
    // B 32x16 layout (same contiguous k order as A): lane<16 holds N=lane, K=0..15; lane>=16 K=16..31.
    const bf16_t* bp = Bb + (long)(n0 + r) * K + half * 16;
    const long    bSub = 16L * K;                 // elements between N-subtiles

    v8f acc0 = {}, acc1 = {}, acc2 = {}, acc3 = {};

    for (int k0 = 0; k0 < K; k0 += 32) {
        // ---- issue ALL loads of this k-step first (2 A + 8 B b128 loads) ----
        v8bf a_lo = *(const v8bf*)(ap + k0);              // K += {0..7}  (+half*16)
        v8bf a_hi = *(const v8bf*)(ap + k0 + 8);          // K += {8..15} (+half*16)
        v8bf bl0  = *(const v8bf*)(bp + 0 * bSub + k0);
        v8bf bh0  = *(const v8bf*)(bp + 0 * bSub + k0 + 8);
        v8bf bl1  = *(const v8bf*)(bp + 1 * bSub + k0);
        v8bf bh1  = *(const v8bf*)(bp + 1 * bSub + k0 + 8);
        v8bf bl2  = *(const v8bf*)(bp + 2 * bSub + k0);
        v8bf bh2  = *(const v8bf*)(bp + 2 * bSub + k0 + 8);
        v8bf bl3  = *(const v8bf*)(bp + 3 * bSub + k0);
        v8bf bh3  = *(const v8bf*)(bp + 3 * bSub + k0 + 8);

        // prefetch next k-step of the B stream (global_prefetch_b8)
        if (k0 + 32 < K) __builtin_prefetch(bp + k0 + 32, 0, 1);

        v16bf a  = frag16(a_lo, a_hi);
        v16bf b0 = frag16(bl0, bh0);
        v16bf b1 = frag16(bl1, bh1);
        v16bf b2 = frag16(bl2, bh2);
        v16bf b3 = frag16(bl3, bh3);

        // ---- 4 independent WMMAs sharing the A fragment ----
        acc0 = __builtin_amdgcn_wmma_f32_16x16x32_bf16(false, a, false, b0, (short)0, acc0, false, false);
        acc1 = __builtin_amdgcn_wmma_f32_16x16x32_bf16(false, a, false, b1, (short)0, acc1, false, false);
        acc2 = __builtin_amdgcn_wmma_f32_16x16x32_bf16(false, a, false, b2, (short)0, acc2, false, false);
        acc3 = __builtin_amdgcn_wmma_f32_16x16x32_bf16(false, a, false, b3, (short)0, acc3, false, false);
    }

    // C/D layout: VGPR v -> M = m0 + half*8 + v, N = n0 + u*16 + r
#pragma unroll
    for (int v = 0; v < 8; ++v) {
        int m = m0 + half * 8 + v;
        float bv = bias[m];
        float* row = Cb + (long)m * N + n0 + r;
        row[0]  = acc0[v] + bv;
        row[16] = acc1[v] + bv;
        row[32] = acc2[v] + bv;
        row[48] = acc3[v] + bv;
    }
}

// ---------------------------------------------------------------------------
// 9-tap dilated local attention. One thread per (b, dilation i, head h, pixel l).
// qkv layout: (B, 768, L) with o = s*256 + i*64 + h*32 + c. Zero padding =>
// out-of-bounds taps contribute logit 0 (still in softmax) and v = 0.
// stacked out: (B, L, ND, 64) f32.
// ---------------------------------------------------------------------------
__global__ void k_attn(const float* __restrict__ qkv, float* __restrict__ stacked) {
    int idx = blockIdx.x * 256 + threadIdx.x;   // over 8*4*2*4096 = 262,144
    int l = idx & (L_PIX - 1);
    int t = idx >> 12;
    int h = t & 1;  t >>= 1;
    int i = t & 3;  int b = t >> 2;
    int dil = 1 << i;
    int y = l >> 6, x = l & 63;

    const float* base = qkv + ((long)b * 768 + i * 64 + h * 32) * L_PIX;
    const float* qp = base;
    const float* kp = base + 256L * L_PIX;
    const float* vp = base + 512L * L_PIX;

    float q[32];
#pragma unroll
    for (int c = 0; c < 32; ++c) q[c] = qp[(long)c * L_PIX + l];

    float logit[9];
    int   pix[9];
#pragma unroll
    for (int j = 0; j < 9; ++j) {
        int yy = y + dil * ((j / 3) - 1);
        int xx = x + dil * ((j % 3) - 1);
        bool ok = (yy >= 0) && (yy < 64) && (xx >= 0) && (xx < 64);
        int p = yy * 64 + xx;
        pix[j] = ok ? p : -1;
        float d = 0.0f;
        if (ok) {
#pragma unroll
            for (int c = 0; c < 32; ++c) d += q[c] * kp[(long)c * L_PIX + p];
            d *= SCALE_F;
        }
        logit[j] = d;   // zero-padded k => exact 0 logit for OOB taps
    }

    float m = logit[0];
#pragma unroll
    for (int j = 1; j < 9; ++j) m = fmaxf(m, logit[j]);
    float w[9], s = 0.0f;
#pragma unroll
    for (int j = 0; j < 9; ++j) { w[j] = __expf(logit[j] - m); s += w[j]; }
    float inv = 1.0f / s;

    float acc[32];
#pragma unroll
    for (int c = 0; c < 32; ++c) acc[c] = 0.0f;
#pragma unroll
    for (int j = 0; j < 9; ++j) {
        if (pix[j] < 0) continue;            // v is zero there
        float wj = w[j] * inv;
#pragma unroll
        for (int c = 0; c < 32; ++c) acc[c] += wj * vp[(long)c * L_PIX + pix[j]];
    }

    float* outp = stacked + (((long)b * L_PIX + l) * 4 + i) * 64 + h * 32;
#pragma unroll
    for (int c = 0; c < 32; ++c) outp[c] = acc[c];
}

// ---------------------------------------------------------------------------
// fc scale-mix (4x4) + residual + LayerNorm over 64 channels per scale.
// One block per pixel; thread = t*64 + c. Output y2 bf16 (B, L, 256),
// K-contiguous, feeding the proj WMMA GEMM directly.
// ---------------------------------------------------------------------------
__global__ void k_fc_ln(const float* __restrict__ stacked,
                        const float* __restrict__ fc_w, const float* __restrict__ fc_b,
                        const float* __restrict__ ln_g, const float* __restrict__ ln_b,
                        bf16_t* __restrict__ y2) {
    __shared__ float sin_[256];
    __shared__ float sf_[256];
    long pix = blockIdx.x;                 // b*L + l
    int tid = threadIdx.x;
    int t = tid >> 6, c = tid & 63;

    sin_[tid] = stacked[pix * 256 + tid];
    __syncthreads();

    float f = fc_b[t] + sin_[t * 64 + c];  // bias + residual
#pragma unroll
    for (int s = 0; s < 4; ++s) f += fc_w[t * 4 + s] * sin_[s * 64 + c];
    sf_[tid] = f;
    __syncthreads();

    float mu = 0.0f, m2 = 0.0f;
    for (int k = 0; k < 64; ++k) {         // broadcast reads within wave
        float v = sf_[t * 64 + k];
        mu += v; m2 += v * v;
    }
    mu *= (1.0f / 64.0f);
    m2 *= (1.0f / 64.0f);
    float var = m2 - mu * mu;
    float nrm = (f - mu) * rsqrtf(var + LN_EPS) * ln_g[c] + ln_b[c];
    y2[pix * 256 + tid] = (bf16_t)nrm;
}

// ---------------------------------------------------------------------------
extern "C" void kernel_launch(void* const* d_in, const int* in_sizes, int n_in,
                              void* d_out, int out_size, void* d_ws, size_t ws_size,
                              hipStream_t stream) {
    (void)in_sizes; (void)n_in; (void)out_size; (void)ws_size;

    const float* x      = (const float*)d_in[0];
    const float* qkv_w  = (const float*)d_in[1];
    const float* qkv_b  = (const float*)d_in[2];
    const float* fc_w   = (const float*)d_in[3];
    const float* fc_b   = (const float*)d_in[4];
    const float* ln_g   = (const float*)d_in[5];
    const float* ln_b   = (const float*)d_in[6];
    const float* proj_w = (const float*)d_in[7];
    const float* proj_b = (const float*)d_in[8];

    char* ws = (char*)d_ws;
    bf16_t* xT       = (bf16_t*)(ws);                 // 16,777,216 B  (B,L,C) bf16
    bf16_t* qkvw_bf  = (bf16_t*)(ws + 16777216);      //    393,216 B  (768,256) bf16
    bf16_t* projw_bf = (bf16_t*)(ws + 17170432);      //    131,072 B  (256,256) bf16
    float*  qkvbuf   = (float* )(ws + 17301504);      // 100,663,296 B (B,768,L) f32
    float*  stacked  = (float* )(ws + 117964800);     //  33,554,432 B (B,L,4,64) f32
    bf16_t* y2       = (bf16_t*)(ws + 151519232);     //  16,777,216 B (B,L,256) bf16
    float*  out      = (float*)d_out;                 // (B,256,L) f32

    // 1) activation transpose + bf16 convert; weight converts
    k_x_to_bf16T<<<32768, 256, 0, stream>>>(x, xT);
    k_f32_to_bf16<<<768, 256, 0, stream>>>(qkv_w, qkvw_bf, 768 * 256);
    k_f32_to_bf16<<<256, 256, 0, stream>>>(proj_w, projw_bf, 256 * 256);

    // 2) qkv GEMM: M=768, K=256, N=4096 per batch (WMMA bf16)
    k_gemm_bf16<<<dim3(8, 48, BATCH), 256, 0, stream>>>(
        qkvw_bf, xT, qkv_b, qkvbuf, 768, L_PIX, C_DIM,
        (long)L_PIX * C_DIM, (long)768 * L_PIX);

    // 3) multi-dilation 9-tap local attention
    k_attn<<<1024, 256, 0, stream>>>(qkvbuf, stacked);

    // 4) fc mix + residual + LayerNorm -> bf16 activations
    k_fc_ln<<<32768, 256, 0, stream>>>(stacked, fc_w, fc_b, ln_g, ln_b, y2);

    // 5) proj GEMM: M=256, K=256, N=4096 per batch -> d_out (B,C,H,W)
    k_gemm_bf16<<<dim3(8, 16, BATCH), 256, 0, stream>>>(
        projw_bf, y2, proj_b, out, 256, L_PIX, C_DIM,
        (long)L_PIX * C_DIM, (long)256 * L_PIX);
}
